// MultiHeadAttention_13872744366838
// MI455X (gfx1250) — compile-verified, matrix operand order checked by reading, not run on hardware
//
#include <hip/hip_runtime.h>

// ---------------------------------------------------------------------------
// MI455X (gfx1250) fused multi-head attention, bf16 WMMA pipeline.
//   x[B,L,D] --cvt--> bf16
//   qkv = x @ qkv_w^T + qkv_b            (WMMA GEMM, TDM-staged, bf16 out)
//   ctx = softmax(QK^T * scale + mask) V (fused flash-style WMMA kernel)
//   out = ctx @ proj_w^T + proj_b        (WMMA GEMM, TDM-staged, fp32 out)
// ~69 GFLOP vs ~50MB HBM traffic -> compute-bound -> v_wmma_f32_16x16x32_bf16
// everywhere; GEMM tiles DMA'd by the Tensor Data Mover with descriptor
// padding replicating the bank-conflict-avoiding LDS row stride.
// ---------------------------------------------------------------------------

typedef __bf16 bf16_t;
typedef __attribute__((ext_vector_type(16))) __bf16 v16bf;
typedef __attribute__((ext_vector_type(8)))  __bf16 v8bf;
typedef __attribute__((ext_vector_type(8)))  float  v8f;
typedef __attribute__((ext_vector_type(4)))  unsigned int v4u;
typedef __attribute__((ext_vector_type(8)))  int v8i;
typedef __attribute__((ext_vector_type(4)))  int v4i;

#define DIM   1024
#define NH    16
#define HD    64
#define SEQ   2048
#define BATCH 2
#define SCALE 0.125f   // HD^-0.5

__device__ __forceinline__ v8f vzero8() {
  v8f z;
#pragma unroll
  for (int i = 0; i < 8; ++i) z[i] = 0.0f;
  return z;
}

// Load a 16x32 bf16 WMMA A-fragment (or 32x16 B-fragment) from row-major LDS.
// ISA layout (16-bit A, wave32): lanes 0-15 row=lane, elems 0..7 = K0..7,
// elems 8..15 = K16..23; lanes 16-31 row=lane-16, K8..15 / K24..31.
// stride must be a multiple of 8 elements (16B) for aligned ds_load_b128.
__device__ __forceinline__ v16bf load_frag16(const bf16_t* base, int stride) {
  const int lane = threadIdx.x & 31;
  const int r    = lane & 15;
  const int half = lane >> 4;
  const bf16_t* p = base + r * stride + half * 8;
  v8bf lo = *(const v8bf*)(p);
  v8bf hi = *(const v8bf*)(p + 16);
  v16bf f;
#pragma unroll
  for (int i = 0; i < 8; ++i) { f[i] = lo[i]; f[i + 8] = hi[i]; }
  return f;
}

__device__ __forceinline__ v8f wmma_bf16(v16bf a, v16bf b, v8f c) {
  return __builtin_amdgcn_wmma_f32_16x16x32_bf16(false, a, false, b, (short)0, c,
                                                 false, false);
}

// ---------------------------------------------------------------------------
// TDM: DMA a 2D bf16 tile [rows x tile_x] (row stride = stride_elems) from
// global memory into LDS at lds_byte_addr, inserting pad_amount DWORDs of LDS
// padding every (pad_interval) DWORDs (codes per D# group1 encoding).
// Descriptor packing per CDNA5 ISA.
// Issued by one wave; completion tracked by that wave's TENSORcnt.
// This toolchain's builtin is the 6-arg form:
//   (v4u g0, v8i g1, v4i g2, v4i g3, v8i extra, i32 cpol)
// ---------------------------------------------------------------------------
__device__ __forceinline__ void tdm_load_2d_bf16(unsigned lds_byte_addr,
                                                 const bf16_t* gptr,
                                                 unsigned tile_x, unsigned rows,
                                                 unsigned stride_elems,
                                                 unsigned pad_i_code,
                                                 unsigned pad_a_code) {
  unsigned long long ga = (unsigned long long)(size_t)gptr;
  v4u g0;
  g0[0] = 1u;                                   // count=1, user mode
  g0[1] = lds_byte_addr;                        // lds_addr [63:32]
  g0[2] = (unsigned)(ga & 0xffffffffu);         // global_addr low
  g0[3] = (unsigned)((ga >> 32) & 0x01ffffffu)  // global_addr [56:32]
          | (2u << 30);                         // type = 2 ("image")
  unsigned td0 = stride_elems;                  // tensor_dim0 (never OOB)
  unsigned td1 = 1u << 20;                      // tensor_dim1 (plenty of rows)
  v8i g1;
  g1[0] = (int)((1u << 16)                      // data_size = 1 (2 bytes)
                | (1u << 20)                    // pad_enable
                | (pad_i_code << 22)            // pad_interval
                | (pad_a_code << 25));          // pad_amount
  g1[1] = (int)((td0 & 0xffffu) << 16);         // [47:32] atomic addr=0, td0 lo
  g1[2] = (int)(((td0 >> 16) & 0xffffu) | ((td1 & 0xffffu) << 16));
  g1[3] = (int)(((td1 >> 16) & 0xffffu) | (tile_x << 16));   // tile_dim0
  g1[4] = (int)(rows & 0xffffu);                // tile_dim1; tile_dim2 = 0
  unsigned long long s0 = (unsigned long long)stride_elems;  // dim0 stride
  g1[5] = (int)(s0 & 0xffffffffu);
  g1[6] = (int)((s0 >> 32) & 0xffffu);          // dim1 stride = 0
  g1[7] = 0;
  v4i z4 = {0, 0, 0, 0};                        // 2D tensor: groups 2/3 unused
  v8i z8 = {0, 0, 0, 0, 0, 0, 0, 0};            // unused extra group
  __builtin_amdgcn_tensor_load_to_lds(g0, g1, z4, z4, z8, 0);
}

// ---------------------------------------------------------------------------
// fp32 -> bf16 conversion (grid-stride)
// ---------------------------------------------------------------------------
__global__ void cvt_f32_bf16(const float* __restrict__ in,
                             bf16_t* __restrict__ out, int n) {
  int i = blockIdx.x * blockDim.x + threadIdx.x;
  int stride = gridDim.x * blockDim.x;
  for (; i < n; i += stride) out[i] = (bf16_t)in[i];
}

// ---------------------------------------------------------------------------
// C[M,N] = A[M,K] @ W[N,K]^T + bias[N]   (torch Linear), TDM double-buffered.
// 128x128 block tile, TK=32, 8 waves; wave = 32x64 sub-tile (2x4 WMMA accs).
// Waves 0/1 issue tensor_load_to_lds for tile k+1 into the spare buffer while
// all 8 waves run WMMA on tile k; readiness = s_wait_tensorcnt + barrier.
// ---------------------------------------------------------------------------
template <bool OUT_BF16>
__global__ __launch_bounds__(256) void gemm_wmma_bf16(
    const bf16_t* __restrict__ A, const bf16_t* __restrict__ W,
    const float* __restrict__ bias, void* __restrict__ Cout,
    int M, int N, int K) {
  constexpr int TM = 128, TN = 128, TK = 32, LDT = TK + 8;  // 40*2B=80B rows
  __shared__ __align__(16) bf16_t lsA[2][TM * LDT];
  __shared__ __align__(16) bf16_t lsB[2][TN * LDT];

  const int tid   = threadIdx.x;
  const int wave  = tid >> 5;
  const int lane  = tid & 31;
  const int waveM = wave & 3;   // 4 waves along M (32 rows each)
  const int waveN = wave >> 2;  // 2 waves along N (64 cols each)
  const int bm = blockIdx.y * TM;
  const int bn = blockIdx.x * TN;

  // wave-relative LDS byte addresses for the TDM descriptors
  const unsigned aoff[2] = {(unsigned)(size_t)&lsA[0][0],
                            (unsigned)(size_t)&lsA[1][0]};
  const unsigned boff[2] = {(unsigned)(size_t)&lsB[0][0],
                            (unsigned)(size_t)&lsB[1][0]};

  v8f acc[2][4];
#pragma unroll
  for (int i = 0; i < 2; ++i)
#pragma unroll
    for (int j = 0; j < 4; ++j) acc[i][j] = vzero8();

  // Prologue: DMA tile 0 into buffer 0.
  // pad codes: row = 32 bf16 = 16 DWORDs -> interval code 3; pad = 8 bf16 =
  // 4 DWORDs -> amount code 3  ==> LDS row stride 40 bf16 == LDT.
  if (wave == 0)
    tdm_load_2d_bf16(aoff[0], A + (size_t)bm * K, TK, TM, K, 3, 3);
  if (wave == 1)
    tdm_load_2d_bf16(boff[0], W + (size_t)bn * K, TK, TN, K, 3, 3);

  int buf = 0;
  for (int k0 = 0; k0 < K; k0 += TK) {
    if (wave < 2) __builtin_amdgcn_s_wait_tensorcnt(0);  // issuer's own cnt
    __syncthreads();                                     // tile `buf` ready
    if (k0 + TK < K) {  // overlap: DMA next tile into the spare buffer
      if (wave == 0)
        tdm_load_2d_bf16(aoff[buf ^ 1], A + (size_t)bm * K + k0 + TK, TK, TM, K, 3, 3);
      if (wave == 1)
        tdm_load_2d_bf16(boff[buf ^ 1], W + (size_t)bn * K + k0 + TK, TK, TN, K, 3, 3);
    }
    const bf16_t* cA = lsA[buf];
    const bf16_t* cB = lsB[buf];
    v16bf af[2], bf[4];
#pragma unroll
    for (int i = 0; i < 2; ++i)
      af[i] = load_frag16(cA + (waveM * 32 + i * 16) * LDT, LDT);
#pragma unroll
    for (int j = 0; j < 4; ++j)
      bf[j] = load_frag16(cB + (waveN * 64 + j * 16) * LDT, LDT);
#pragma unroll
    for (int i = 0; i < 2; ++i)
#pragma unroll
      for (int j = 0; j < 4; ++j) acc[i][j] = wmma_bf16(af[i], bf[j], acc[i][j]);
    buf ^= 1;
  }

  // C/D layout: elem e -> row = e + 8*(lane>=16), col = lane&15.
  const int rlo = (lane >> 4) << 3;
  const int cl  = lane & 15;
#pragma unroll
  for (int i = 0; i < 2; ++i)
#pragma unroll
    for (int j = 0; j < 4; ++j) {
      int col = bn + waveN * 64 + j * 16 + cl;
      float bv = bias[col];
#pragma unroll
      for (int e = 0; e < 8; ++e) {
        int row = bm + waveM * 32 + i * 16 + rlo + e;
        float v = acc[i][j][e] + bv;
        if constexpr (OUT_BF16)
          ((bf16_t*)Cout)[(size_t)row * N + col] = (bf16_t)v;
        else
          ((float*)Cout)[(size_t)row * N + col] = v;
      }
    }
}

// ---------------------------------------------------------------------------
// Fused attention: one block per (128-query tile, head, batch); 8 waves,
// each wave owns 16 query rows. Streams 64-key tiles with online softmax.
// qkv rows are [B*L, 3*DIM]; Q at col h*64, K at DIM+h*64, V at 2*DIM+h*64.
// ---------------------------------------------------------------------------
__global__ __launch_bounds__(256) void attn_wmma(
    const bf16_t* __restrict__ qkv, const int* __restrict__ mask,
    bf16_t* __restrict__ ctx) {
  constexpr int QT = 128, KT = 64, LQ = HD + 8 /*72*/, LP = KT + 8 /*72*/;
  __shared__ __align__(16) bf16_t lq[QT * LQ];        // Q tile   (18 KB)
  __shared__ __align__(16) bf16_t lk[KT * LQ];        // K tile   ( 9 KB)
  __shared__ __align__(16) bf16_t lvt[HD * LP];       // V^T tile ( 9 KB)
  __shared__ __align__(16) bf16_t lp[8 * 16 * LP];    // per-wave P (18 KB)

  const int qb = blockIdx.x, h = blockIdx.y, b = blockIdx.z;
  const int tid = threadIdx.x, wave = tid >> 5, lane = tid & 31;
  const size_t row0 = (size_t)(b * SEQ) * (3 * DIM);
  const int qcol = h * HD, kcol = DIM + h * HD, vcol = 2 * DIM + h * HD;

  // ---- stage Q tile (128 x 64) ----
  for (int c = tid; c < QT * (HD / 8); c += 256) {
    int r = c >> 3, d8 = (c & 7) * 8;
    *(v8bf*)(lq + r * LQ + d8) =
        *(const v8bf*)(qkv + row0 + (size_t)(qb * QT + r) * (3 * DIM) + qcol + d8);
  }
  __syncthreads();

  v16bf qf0 = load_frag16(lq + (wave * 16) * LQ, LQ);       // hd 0..31
  v16bf qf1 = load_frag16(lq + (wave * 16) * LQ + 32, LQ);  // hd 32..63

  v8f o[4];
#pragma unroll
  for (int j = 0; j < 4; ++j) o[j] = vzero8();
  float mrun[8], lrun[8];
#pragma unroll
  for (int e = 0; e < 8; ++e) { mrun[e] = -1e30f; lrun[e] = 0.0f; }

  const int rlo = (lane >> 4) << 3;
  const int cl  = lane & 15;
  const int qrow_g = qb * QT + wave * 16 + rlo;  // + e => query row in [0,SEQ)

  for (int kt = 0; kt < SEQ / KT; ++kt) {
    asm volatile("s_wait_dscnt 0" ::: "memory");
    __syncthreads();
    // ---- stage K tile (64 x 64, row-major) ----
    for (int c = tid; c < KT * (HD / 8); c += 256) {
      int r = c >> 3, d8 = (c & 7) * 8;
      *(v8bf*)(lk + r * LQ + d8) =
          *(const v8bf*)(qkv + row0 + (size_t)(kt * KT + r) * (3 * DIM) + kcol + d8);
    }
    // ---- stage V transposed: lvt[d][key] ----
    for (int c = tid; c < KT * (HD / 8); c += 256) {
      int r = c >> 3, d8 = (c & 7) * 8;
      v8bf v = *(const v8bf*)(qkv + row0 + (size_t)(kt * KT + r) * (3 * DIM) + vcol + d8);
#pragma unroll
      for (int e = 0; e < 8; ++e) lvt[(d8 + e) * LP + r] = v[e];
    }
    // pull next iteration's K/V rows toward L2 (global_prefetch_b8)
    if (kt + 1 < SEQ / KT && tid < 128) {
      int r = tid & 63;
      __builtin_prefetch(qkv + row0 + (size_t)((kt + 1) * KT + r) * (3 * DIM) +
                             (tid < 64 ? kcol : vcol), 0, 1);
    }
    __syncthreads();

    // ---- S = Q K^T   (4 n-tiles x 2 k-steps = 8 WMMA) ----
    v8f s[4];
#pragma unroll
    for (int n = 0; n < 4; ++n) {
      s[n] = vzero8();
      v16bf kf0 = load_frag16(lk + (n * 16) * LQ, LQ);
      v16bf kf1 = load_frag16(lk + (n * 16) * LQ + 32, LQ);
      s[n] = wmma_bf16(qf0, kf0, s[n]);
      s[n] = wmma_bf16(qf1, kf1, s[n]);
    }

    // ---- scale + mask + online softmax ----
    float mloc[8];
#pragma unroll
    for (int e = 0; e < 8; ++e) mloc[e] = -1e30f;
#pragma unroll
    for (int n = 0; n < 4; ++n) {
      int kg = kt * KT + n * 16 + cl;
#pragma unroll
      for (int e = 0; e < 8; ++e) {
        float sv = s[n][e] * SCALE;
        if (mask[(size_t)(qrow_g + e) * SEQ + kg] == 0) sv = -1e30f;
        s[n][e] = sv;
        mloc[e] = fmaxf(mloc[e], sv);
      }
    }
    // a score row lives across 16 lanes of a half-wave -> xor-reduce 1,2,4,8
#pragma unroll
    for (int e = 0; e < 8; ++e)
#pragma unroll
      for (int off = 1; off < 16; off <<= 1)
        mloc[e] = fmaxf(mloc[e], __shfl_xor(mloc[e], off, 32));

    float alpha[8], rs[8];
#pragma unroll
    for (int e = 0; e < 8; ++e) {
      float mn = fmaxf(mrun[e], mloc[e]);
      alpha[e] = __expf(mrun[e] - mn);
      mrun[e]  = mn;
      rs[e]    = 0.0f;
    }
#pragma unroll
    for (int n = 0; n < 4; ++n)
#pragma unroll
      for (int e = 0; e < 8; ++e) {
        float p = __expf(s[n][e] - mrun[e]);
        s[n][e] = p;
        rs[e] += p;
      }
#pragma unroll
    for (int e = 0; e < 8; ++e) {
#pragma unroll
      for (int off = 1; off < 16; off <<= 1) rs[e] += __shfl_xor(rs[e], off, 32);
      lrun[e] = lrun[e] * alpha[e] + rs[e];
    }
#pragma unroll
    for (int j = 0; j < 4; ++j)
#pragma unroll
      for (int e = 0; e < 8; ++e) o[j][e] *= alpha[e];

    // ---- P: C-layout -> A-layout via per-wave LDS strip ----
    bf16_t* myP = lp + wave * 16 * LP;
#pragma unroll
    for (int n = 0; n < 4; ++n)
#pragma unroll
      for (int e = 0; e < 8; ++e)
        myP[(rlo + e) * LP + n * 16 + cl] = (bf16_t)s[n][e];
    asm volatile("s_wait_dscnt 0" ::: "memory");  // same-wave store->frag-load

    // ---- O += P V   (2 k-steps x 4 hd-tiles = 8 WMMA) ----
#pragma unroll
    for (int kk = 0; kk < 2; ++kk) {
      v16bf pf = load_frag16(myP + kk * 32, LP);
#pragma unroll
      for (int j = 0; j < 4; ++j) {
        v16bf vf = load_frag16(lvt + (j * 16) * LP + kk * 32, LP);
        o[j] = wmma_bf16(pf, vf, o[j]);
      }
    }
  }

  // ---- epilogue: O /= l, write ctx[B*L, DIM] at col h*64 ----
  float invl[8];
#pragma unroll
  for (int e = 0; e < 8; ++e) invl[e] = 1.0f / lrun[e];
#pragma unroll
  for (int j = 0; j < 4; ++j)
#pragma unroll
    for (int e = 0; e < 8; ++e) {
      size_t row = (size_t)b * SEQ + qrow_g + e;
      int col = h * HD + j * 16 + cl;
      ctx[row * DIM + col] = (bf16_t)(o[j][e] * invl[e]);
    }
}

// ---------------------------------------------------------------------------
extern "C" void kernel_launch(void* const* d_in, const int* in_sizes, int n_in,
                              void* d_out, int out_size, void* d_ws, size_t ws_size,
                              hipStream_t stream) {
  const float* x      = (const float*)d_in[0];
  const int*   mask   = (const int*)d_in[1];
  const float* qkv_w  = (const float*)d_in[2];
  const float* qkv_b  = (const float*)d_in[3];
  const float* proj_w = (const float*)d_in[4];
  const float* proj_b = (const float*)d_in[5];
  float* out = (float*)d_out;

  const int ROWS = BATCH * SEQ;  // 4096
  char* ws = (char*)d_ws;
  size_t off = 0;
  bf16_t* xb    = (bf16_t*)(ws + off); off += (size_t)ROWS * DIM * 2;        // 8 MB
  bf16_t* wqkv  = (bf16_t*)(ws + off); off += (size_t)3 * DIM * DIM * 2;     // 6 MB
  bf16_t* wproj = (bf16_t*)(ws + off); off += (size_t)DIM * DIM * 2;         // 2 MB
  bf16_t* qkvb  = (bf16_t*)(ws + off); off += (size_t)ROWS * 3 * DIM * 2;    // 24 MB
  bf16_t* ctxb  = (bf16_t*)(ws + off); off += (size_t)ROWS * DIM * 2;        // 8 MB

  // fp32 -> bf16
  cvt_f32_bf16<<<dim3(4096), dim3(256), 0, stream>>>(x, xb, ROWS * DIM);
  cvt_f32_bf16<<<dim3(4096), dim3(256), 0, stream>>>(qkv_w, wqkv, 3 * DIM * DIM);
  cvt_f32_bf16<<<dim3(2048), dim3(256), 0, stream>>>(proj_w, wproj, DIM * DIM);

  // qkv = x @ qkv_w^T + qkv_b   -> bf16 [4096, 3072]
  gemm_wmma_bf16<true><<<dim3(3 * DIM / 128, ROWS / 128), dim3(256), 0, stream>>>(
      xb, wqkv, qkv_b, (void*)qkvb, ROWS, 3 * DIM, DIM);

  // fused attention -> ctx bf16 [4096, 1024]
  attn_wmma<<<dim3(SEQ / 128, NH, BATCH), dim3(256), 0, stream>>>(qkvb, mask, ctxb);

  // out = ctx @ proj_w^T + proj_b -> fp32 [4096, 1024]
  gemm_wmma_bf16<false><<<dim3(DIM / 128, ROWS / 128), dim3(256), 0, stream>>>(
      ctxb, wproj, proj_b, (void*)out, ROWS, DIM, DIM);
}
